// TraceAxisAttention2D_gate_29678224015613
// MI455X (gfx1250) — matrix operand order repaired and checked by reading, not compiled, not run on hardware
//
#include <hip/hip_runtime.h>

// ---------------------------------------------------------------------------
// TraceAxisAttention2D_gate  — CDNA5 (gfx1250) WMMA implementation
// B=2, H=512, W=64, C=256, NH=8, hd=32, half=16.  Rows R = bw*512 + h,
// bw = b*64 + w.  All GEMMs via v_wmma_f32_16x16x32_f16 (f32 accum).
// ---------------------------------------------------------------------------

typedef __attribute__((ext_vector_type(16))) _Float16 v16h;
typedef __attribute__((ext_vector_type(8)))  _Float16 v8h;
typedef __attribute__((ext_vector_type(8)))  float    v8f;
typedef int vsi4 __attribute__((vector_size(4 * sizeof(int))));  // builtin arg type

#define ROWS   65536   // B*W*H = 128*512
#define CC     256
#define NHEAD  8
#define HD     32

#if defined(__gfx1250__) && __has_builtin(__builtin_amdgcn_global_load_async_to_lds_b128)
#define HAVE_ASYNC_LDS 1
#else
#define HAVE_ASYNC_LDS 0
#endif

__device__ __forceinline__ void wait_async0() {
#if defined(__gfx1250__)
#if __has_builtin(__builtin_amdgcn_s_wait_asynccnt)
  __builtin_amdgcn_s_wait_asynccnt(0);
#else
  asm volatile("s_wait_asynccnt 0" ::: "memory");
#endif
#endif
}

// ---- WMMA fragment loaders (wave32 layouts per CDNA5 ISA 7.12.2) ----------
// A (16x32 f16): lane L: M = L%16;  element e -> K = (e%8) + 8*((e/8)*2 + L/16)
__device__ __forceinline__ v16h ldA(const _Float16* base, int ld) {
  int lane = threadIdx.x & 31;
  int m = lane & 15;
  int hi = lane >> 4;
  const _Float16* row = base + m * ld;
  v8h lo = *(const v8h*)(row + 8 * hi);
  v8h hv = *(const v8h*)(row + 16 + 8 * hi);
  v16h r;
#pragma unroll
  for (int e = 0; e < 8; ++e) { r[e] = lo[e]; r[8 + e] = hv[e]; }
  return r;
}

// B (32x16 f16) loaded from an N-major (K-contiguous) buffer:
// lane L: N = L%16; element e -> K = (e%8) + 8*((e/8)*2 + L/16)
__device__ __forceinline__ v16h ldB(const _Float16* base, int ld) {
  int lane = threadIdx.x & 31;
  int nn = lane & 15;
  int hi = lane >> 4;
  const _Float16* row = base + nn * ld + 8 * hi;
  v8h a = *(const v8h*)(row);
  v8h b = *(const v8h*)(row + 16);
  v16h r;
#pragma unroll
  for (int e = 0; e < 8; ++e) { r[e] = a[e]; r[8 + e] = b[e]; }
  return r;
}

__device__ __forceinline__ v8f wmma16(v16h a, v16h b, v8f c) {
  return __builtin_amdgcn_wmma_f32_16x16x32_f16(false, a, false, b,
                                                (short)0, c, false, false);
}

// ---------------------------------------------------------------------------
// K0: weight transposes -> f16 (K-contiguous "B" layout) + RoPE tables
// ---------------------------------------------------------------------------
__global__ void k_prep(const float* __restrict__ Wq, const float* __restrict__ Wk,
                       const float* __restrict__ Wv, const float* __restrict__ Wo,
                       const float* __restrict__ pos, const float* __restrict__ rope_w,
                       const float* __restrict__ rope_b,
                       _Float16* __restrict__ Wt, _Float16* __restrict__ Wot,
                       float* __restrict__ cosT, float* __restrict__ sinT) {
  int tid = blockIdx.x * blockDim.x + threadIdx.x;
  int stride = gridDim.x * blockDim.x;
  // Wq (256 x 512) -> Wt rows [0,512)
  for (int i = tid; i < 512 * 256; i += stride) {
    int f = i >> 8, c = i & 255;
    Wt[f * 256 + c] = (_Float16)Wq[c * 512 + f];
  }
  // Wk, Wv -> Wt rows [512,768) and [768,1024); Wo -> Wot
  for (int i = tid; i < 256 * 256; i += stride) {
    int f = i >> 8, c = i & 255;
    Wt[(512 + f) * 256 + c] = (_Float16)Wk[c * 256 + f];
    Wt[(768 + f) * 256 + c] = (_Float16)Wv[c * 256 + f];
    Wot[f * 256 + c]        = (_Float16)Wo[c * 256 + f];
  }
  // RoPE tables: cos/sin[(b*512+h)*128 + n*16 + j]
  for (int i = tid; i < 2 * 512 * 128; i += stride) {
    int j = i & 127;
    int h = (i >> 7) & 511;
    int b = i >> 16;
    const float* pp = pos + (size_t)(b * 512 + h) * 4;
    float p = pp[0] * rope_w[0] + pp[1] * rope_w[1] +
              pp[2] * rope_w[2] + pp[3] * rope_w[3] + rope_b[0];
    float freq = __expf(-12.0f + 12.0f * (float)j / 127.0f);
    float ang = p * freq;              // P_SCALE = 1
    cosT[i] = __cosf(ang);
    sinT[i] = __sinf(ang);
  }
}

// ---------------------------------------------------------------------------
// K1: fused QKV projection GEMM  Y = X @ [Wq | Wk | Wv]   (65536 x 1024)
//     64 rows / workgroup, X tile staged in LDS as f16, weights streamed
//     from L2.  V written transposed (vt[bw][n][d][h]) for the P·V WMMA.
// ---------------------------------------------------------------------------
__global__ void __launch_bounds__(256) k_proj(const float* __restrict__ x,
                                              const _Float16* __restrict__ Wt,
                                              _Float16* __restrict__ qg,
                                              _Float16* __restrict__ kraw,
                                              _Float16* __restrict__ vt) {
  __shared__ _Float16 sX[64 * 256];
  int wg = blockIdx.x;
  int t = threadIdx.x;
  for (int j = 0; j < 64; ++j) {
    int flat = j * 256 + t;
    int r = flat >> 8, c = flat & 255;
    int R = wg * 64 + r;
    int bw = R >> 9, h = R & 511;
    int b = bw >> 6, w = bw & 63;
    sX[r * 256 + c] = (_Float16)x[(((size_t)(b * 512 + h) * 64 + w) * 256) + c];
  }
  __syncthreads();

  int wave = t >> 5;
  int lane = t & 31;
  int nn = lane & 15, mb = (lane >> 4) * 8;

  for (int tile = wave; tile < 4 * 64; tile += 8) {
    int mt = tile >> 6;      // 0..3  (16-row subtile)
    int nt = tile & 63;      // 0..63 (16-col tile of the 1024 outputs)
    const _Float16* a0 = sX + mt * 16 * 256;
    const _Float16* b0 = Wt + (size_t)nt * 16 * 256;
    __builtin_prefetch(Wt + (size_t)((tile + 8) & 63) * 16 * 256, 0, 1);
    v8f acc = {};
#pragma unroll
    for (int k0 = 0; k0 < 8; ++k0) {
      v16h a = ldA(a0 + k0 * 32, 256);
      v16h b = ldB(b0 + k0 * 32, 256);
      acc = wmma16(a, b, acc);
    }
    int Rbase = wg * 64 + mt * 16 + mb;
    int fbase = nt * 16 + nn;           // global output feature 0..1023
    if (nt < 32) {                      // q | gate -> qg[R][512]
#pragma unroll
      for (int r = 0; r < 8; ++r)
        qg[(size_t)(Rbase + r) * 512 + fbase] = (_Float16)acc[r];
    } else if (nt < 48) {               // k -> kraw[R][256]
      int f = fbase - 512;
#pragma unroll
      for (int r = 0; r < 8; ++r)
        kraw[(size_t)(Rbase + r) * 256 + f] = (_Float16)acc[r];
    } else {                            // v -> vt[bw][n][d][512] (transposed)
      int f = fbase - 768;
      int nh_ = f >> 5, d = f & 31;
#pragma unroll
      for (int r = 0; r < 8; ++r) {
        int R = Rbase + r;
        int bw = R >> 9, h = R & 511;
        vt[((size_t)(bw * 8 + nh_) * 32 + d) * 512 + h] = (_Float16)acc[r];
      }
    }
  }
}

// ---------------------------------------------------------------------------
// K2: per-(row, head) RMSNorm + RoPE for q and k -> qb/kb [bw][n][h][32] f16
// ---------------------------------------------------------------------------
__global__ void __launch_bounds__(256) k_normrope(const _Float16* __restrict__ qg,
                                                  const _Float16* __restrict__ kraw,
                                                  const float* __restrict__ qw,
                                                  const float* __restrict__ kw,
                                                  const float* __restrict__ cosT,
                                                  const float* __restrict__ sinT,
                                                  _Float16* __restrict__ qb,
                                                  _Float16* __restrict__ kb) {
  int i = blockIdx.x * blockDim.x + threadIdx.x;
  if (i >= ROWS * NHEAD) return;
  int n = i & 7;
  int R = i >> 3;
  int bw = R >> 9, h = R & 511;
  int b = bw >> 6;

  const _Float16* qp = qg + (size_t)R * 512 + n * 32;
  const _Float16* kp = kraw + (size_t)R * 256 + n * 32;
  float qv[32], kv[32];
  float qs = 0.f, ks = 0.f;
#pragma unroll
  for (int d = 0; d < 32; ++d) {
    qv[d] = (float)qp[d]; qs += qv[d] * qv[d];
    kv[d] = (float)kp[d]; ks += kv[d] * kv[d];
  }
  float qr = rsqrtf(qs * (1.0f / 32.0f) + 1e-6f);
  float kr = rsqrtf(ks * (1.0f / 32.0f) + 1e-6f);

  const float* ct = cosT + ((size_t)(b * 512 + h)) * 128 + n * 16;
  const float* st = sinT + ((size_t)(b * 512 + h)) * 128 + n * 16;
  _Float16* qo = qb + (((size_t)(bw * 8 + n) * 512) + h) * 32;
  _Float16* ko = kb + (((size_t)(bw * 8 + n) * 512) + h) * 32;
#pragma unroll
  for (int j = 0; j < 16; ++j) {
    float c = ct[j], s = st[j];
    float qe = qw[2 * j] * qv[2 * j] * qr;
    float qo_ = qw[2 * j + 1] * qv[2 * j + 1] * qr;
    qo[2 * j]     = (_Float16)(qe * c - qo_ * s);
    qo[2 * j + 1] = (_Float16)(qe * s + qo_ * c);
    float ke = kw[2 * j] * kv[2 * j] * kr;
    float ko_ = kw[2 * j + 1] * kv[2 * j + 1] * kr;
    ko[2 * j]     = (_Float16)(ke * c - ko_ * s);
    ko[2 * j + 1] = (_Float16)(ke * s + ko_ * c);
  }
}

// ---------------------------------------------------------------------------
// K3: flash attention, one workgroup per (bw, head).
//     Whole head (Q,K,Vt = 96 KB) resident in the 320 KB WGP LDS.
//     Staging uses GLOBAL_LOAD_ASYNC_TO_LDS_B128 (ASYNCcnt) when available.
//     S tile = ONE wmma (K-dim = hd = 32); online softmax; P·V via WMMA
//     through a per-wave LDS P-scratch (C-layout -> A-layout shuffle).
// ---------------------------------------------------------------------------
__global__ void __launch_bounds__(256) k_attn(const _Float16* __restrict__ qb,
                                              const _Float16* __restrict__ kb,
                                              const _Float16* __restrict__ vt,
                                              _Float16* __restrict__ att) {
  extern __shared__ _Float16 smem[];
  _Float16* sQ = smem;               // 512*32
  _Float16* sK = smem + 16384;       // 512*32
  _Float16* sV = smem + 32768;       // 32*512 (Vt)
  _Float16* sPall = smem + 49152;    // 8 waves * 16*32

  int bn = blockIdx.x;               // bw*8 + head
  int t = threadIdx.x;
  const _Float16* gq = qb + (size_t)bn * 512 * 32;
  const _Float16* gk = kb + (size_t)bn * 512 * 32;
  const _Float16* gv = vt + (size_t)bn * 32 * 512;

#if HAVE_ASYNC_LDS
  // CDNA5 async DMA: memory -> LDS without touching VGPRs; tracked by ASYNCcnt.
  for (int j = t; j < 2048; j += 256) {   // 2048 x 16B per array
    __builtin_amdgcn_global_load_async_to_lds_b128(
        (vsi4*)(gq) + j, (vsi4*)(sQ) + j, 0, 0);
    __builtin_amdgcn_global_load_async_to_lds_b128(
        (vsi4*)(gk) + j, (vsi4*)(sK) + j, 0, 0);
    __builtin_amdgcn_global_load_async_to_lds_b128(
        (vsi4*)(gv) + j, (vsi4*)(sV) + j, 0, 0);
  }
  wait_async0();
#else
  for (int j = t; j < 2048; j += 256) {   // 2048 x v8h per array
    ((v8h*)sQ)[j] = ((const v8h*)gq)[j];
    ((v8h*)sK)[j] = ((const v8h*)gk)[j];
    ((v8h*)sV)[j] = ((const v8h*)gv)[j];
  }
#endif
  __syncthreads();

  int wave = t >> 5, lane = t & 31;
  int nn = lane & 15, mb = (lane >> 4) * 8;
  _Float16* myP = sPall + wave * 512;
  const float scale = 0.17677669529663687f;   // 32^-0.5
  int bw = bn >> 3, nh_ = bn & 7;

  for (int qt = wave; qt < 32; qt += 8) {     // 16-query tiles
    v16h qf = ldA(sQ + qt * 16 * 32, 32);
    v8f acc0 = {}, acc1 = {};
    float m[8], l[8];
#pragma unroll
    for (int r = 0; r < 8; ++r) { m[r] = -3.0e38f; l[r] = 0.f; }

    for (int kt = 0; kt < 16; ++kt) {         // 32 keys per step
      v16h kf0 = ldB(sK + (kt * 32) * 32, 32);
      v16h kf1 = ldB(sK + (kt * 32 + 16) * 32, 32);
      v8f z = {};
      v8f s0 = wmma16(qf, kf0, z);
      v8f s1 = wmma16(qf, kf1, z);
#pragma unroll
      for (int r = 0; r < 8; ++r) {
        float a0 = s0[r] * scale, a1 = s1[r] * scale;
        float rm = fmaxf(a0, a1);
        rm = fmaxf(rm, __shfl_xor(rm, 1, 32));
        rm = fmaxf(rm, __shfl_xor(rm, 2, 32));
        rm = fmaxf(rm, __shfl_xor(rm, 4, 32));
        rm = fmaxf(rm, __shfl_xor(rm, 8, 32));
        float mn = fmaxf(m[r], rm);
        float alpha = __expf(m[r] - mn);
        float p0 = __expf(a0 - mn);
        float p1 = __expf(a1 - mn);
        float rs = p0 + p1;
        rs += __shfl_xor(rs, 1, 32);
        rs += __shfl_xor(rs, 2, 32);
        rs += __shfl_xor(rs, 4, 32);
        rs += __shfl_xor(rs, 8, 32);
        m[r] = mn;
        l[r] = l[r] * alpha + rs;
        acc0[r] *= alpha;
        acc1[r] *= alpha;
        myP[(mb + r) * 32 + nn]      = (_Float16)p0;
        myP[(mb + r) * 32 + 16 + nn] = (_Float16)p1;
      }
      // per-wave LDS is in-order: reload P in A-layout, then P·V
      v16h pf  = ldA(myP, 32);
      v16h vf0 = ldB(sV + kt * 32, 512);              // d = 0..15
      v16h vf1 = ldB(sV + 16 * 512 + kt * 32, 512);   // d = 16..31
      acc0 = wmma16(pf, vf0, acc0);
      acc1 = wmma16(pf, vf1, acc1);
    }
#pragma unroll
    for (int r = 0; r < 8; ++r) {
      float inv = 1.0f / l[r];
      size_t R = (size_t)bw * 512 + qt * 16 + mb + r;
      att[R * 256 + nh_ * 32 + nn]      = (_Float16)(acc0[r] * inv);
      att[R * 256 + nh_ * 32 + 16 + nn] = (_Float16)(acc1[r] * inv);
    }
  }
}

// ---------------------------------------------------------------------------
// K4: gated output projection  out = (att * sigmoid(gate)) @ Wo + bo
//     Gating fused into the LDS A-tile stage; final (B,H,W,C) transpose on
//     the accumulator scatter.
// ---------------------------------------------------------------------------
__global__ void __launch_bounds__(256) k_out(const _Float16* __restrict__ att,
                                             const _Float16* __restrict__ qg,
                                             const _Float16* __restrict__ Wot,
                                             const float* __restrict__ bo,
                                             float* __restrict__ out) {
  __shared__ _Float16 sA[64 * 256];
  int wg = blockIdx.x, t = threadIdx.x;
  for (int j = t; j < 64 * 256; j += 256) {
    int r = j >> 8, c = j & 255;
    size_t R = (size_t)wg * 64 + r;
    float a = (float)att[R * 256 + c];
    float g = (float)qg[R * 512 + 256 + c];
    float sg = 1.0f / (1.0f + __expf(-g));
    sA[r * 256 + c] = (_Float16)(a * sg);
  }
  __syncthreads();

  int wave = t >> 5, lane = t & 31;
  int nn = lane & 15, mb = (lane >> 4) * 8;
  for (int tile = wave; tile < 64; tile += 8) {
    int mt = tile >> 4;     // 0..3
    int nt = tile & 15;     // 0..15
    v8f acc = {};
#pragma unroll
    for (int k0 = 0; k0 < 8; ++k0) {
      v16h a = ldA(sA + mt * 16 * 256 + k0 * 32, 256);
      v16h b = ldB(Wot + nt * 16 * 256 + k0 * 32, 256);
      acc = wmma16(a, b, acc);
    }
    int c = nt * 16 + nn;
    float bias = bo[c];
#pragma unroll
    for (int r = 0; r < 8; ++r) {
      int R = wg * 64 + mt * 16 + mb + r;
      int bw = R >> 9, h = R & 511;
      int b = bw >> 6, w = bw & 63;
      out[(((size_t)(b * 512 + h)) * 64 + w) * 256 + c] = acc[r] + bias;
    }
  }
}

// ---------------------------------------------------------------------------
extern "C" void kernel_launch(void* const* d_in, const int* in_sizes, int n_in,
                              void* d_out, int out_size, void* d_ws, size_t ws_size,
                              hipStream_t stream) {
  (void)in_sizes; (void)n_in; (void)out_size; (void)ws_size;
  const float* x      = (const float*)d_in[0];
  const float* pos    = (const float*)d_in[1];
  const float* Wq     = (const float*)d_in[2];
  const float* Wk     = (const float*)d_in[3];
  const float* Wv     = (const float*)d_in[4];
  const float* Wo     = (const float*)d_in[5];
  const float* bo     = (const float*)d_in[6];
  const float* qnw    = (const float*)d_in[7];
  const float* knw    = (const float*)d_in[8];
  const float* rope_w = (const float*)d_in[9];
  const float* rope_b = (const float*)d_in[10];
  float* out = (float*)d_out;

  char* base = (char*)d_ws;
  size_t off = 0;
  auto nxt = [&](size_t bytes) -> char* {
    char* p = base + off;
    off += (bytes + 255) & ~(size_t)255;
    return p;
  };
  _Float16* Wt   = (_Float16*)nxt((size_t)1024 * 256 * 2);
  _Float16* Wot  = (_Float16*)nxt((size_t)256 * 256 * 2);
  float*    cosT = (float*)nxt((size_t)2 * 512 * 128 * 4);
  float*    sinT = (float*)nxt((size_t)2 * 512 * 128 * 4);
  _Float16* qg   = (_Float16*)nxt((size_t)ROWS * 512 * 2);
  _Float16* kraw = (_Float16*)nxt((size_t)ROWS * 256 * 2);
  _Float16* qb   = (_Float16*)nxt((size_t)ROWS * 256 * 2);
  _Float16* kb   = (_Float16*)nxt((size_t)ROWS * 256 * 2);
  _Float16* vt   = (_Float16*)nxt((size_t)ROWS * 256 * 2);
  _Float16* att  = (_Float16*)nxt((size_t)ROWS * 256 * 2);

  k_prep<<<512, 256, 0, stream>>>(Wq, Wk, Wv, Wo, pos, rope_w, rope_b,
                                  Wt, Wot, cosT, sinT);
  k_proj<<<ROWS / 64, 256, 0, stream>>>(x, Wt, qg, kraw, vt);
  k_normrope<<<(ROWS * NHEAD) / 256, 256, 0, stream>>>(qg, kraw, qnw, knw,
                                                       cosT, sinT, qb, kb);
  size_t smem_attn = (size_t)(49152 + 8 * 512) * sizeof(_Float16); // 104 KB
  k_attn<<<128 * NHEAD, 256, smem_attn, stream>>>(qb, kb, vt, att);
  k_out<<<ROWS / 64, 256, 0, stream>>>(att, qg, Wot, bo, out);
}
